// SparseUNet_32341103739274
// MI455X (gfx1250) — compile-verified, matrix operand order checked by reading, not run on hardware
//
#include <hip/hip_runtime.h>

// ---------------------------------------------------------------------------
// MI455X (gfx1250) sparse 3D U-Net forward, v5.
// Implicit-GEMM convs on v_wmma_f32_16x16x32_bf16 (bf16 A/B, f32 accum).
// Block = 256 threads = 8 waves; tile = 128 voxels (M) x 16 Cout (N).
// Per (tap, K-chunk): branch-free cooperative staging into double-buffered
// LDS (v_cvt_pk_bf16_f32 packed pairs -> ds_store_b128/b32), one barrier per
// K-step, fragments via 4x ds_load_b128, one WMMA per wave per K-step.
// Channel tails use clamped float2 pair loads (all Cin even >= 2; conv0 is
// zero-padded 1->2 channels).  Whole net is L2-resident -> matrix-pipe bound.
// ---------------------------------------------------------------------------

typedef __bf16 bf16_t;
typedef __attribute__((ext_vector_type(2)))  float  v2f;
typedef __attribute__((ext_vector_type(2)))  __bf16 v2bf;
typedef __attribute__((ext_vector_type(8)))  __bf16 v8bf;
typedef __attribute__((ext_vector_type(16))) __bf16 v16bf;
typedef __attribute__((ext_vector_type(8)))  float  v8f;

// Pack two floats into 2xbf16: exactly one v_cvt_pk_bf16_f32 (RNE).
__device__ __forceinline__ unsigned pack2bf(float x, float y) {
    v2f f;
    f[0] = x;
    f[1] = y;
    v2bf b = __builtin_convertvector(f, v2bf);
    return __builtin_bit_cast(unsigned, b);
}

__device__ __forceinline__ int clampi(int v, int lo, int hi) {
    return v < lo ? lo : (v > hi ? hi : v);
}

// Fragment from LDS row (layout [row][k], 32 k per row).  CDNA5 16-bit
// fragment layout: lane half 'hi' holds k in [hi*8,hi*8+8) and
// [16+hi*8, 16+hi*8+8) -> two contiguous 16B ds_load_b128 reads.
__device__ __forceinline__ v16bf frag_from_lds(const bf16_t* rowPtr, int hi) {
    v8bf lo = *(const v8bf*)(rowPtr + hi * 8);
    v8bf up = *(const v8bf*)(rowPtr + 16 + hi * 8);
    return __builtin_shufflevector(lo, up, 0, 1, 2, 3, 4, 5, 6, 7,
                                   8, 9, 10, 11, 12, 13, 14, 15);
}

// ---------------------------------------------------------------------------
// Generic masked conv3d, NDHWC, cubic dims.  KS=3/S=1 => SAME(pad1),
// KS=2/S=2 => VALID, KS=1 => pointwise.  Writes channel slice
// [outCoff, outCoff+Cout) of an outCtot-channel tensor; optional residual.
// Requires Cin even (>=2).
// ---------------------------------------------------------------------------
template <int KS, int STRIDE, bool RELU>
__global__ __launch_bounds__(256) void conv3d_wmma_kernel(
    const float* __restrict__ in, const float* __restrict__ w,
    const float* __restrict__ bias, const float* __restrict__ mask,
    const float* __restrict__ residual, float* __restrict__ out,
    int Din, int Cin, int Dq, int Cout,
    int outCtot, int outCoff, int resCtot, int resCoff)
{
    __shared__ __align__(16) bf16_t ldsA[2][128 * 32];  // [buf][vox][k]
    __shared__ __align__(16) bf16_t ldsB[2][16 * 32];   // [buf][n][k]

    const int tid  = threadIdx.x;
    const int lane = tid & 31;
    const int wave = tid >> 5;
    const int hi   = lane >> 4;
    const int lr   = lane & 15;
    const int blockM = blockIdx.x * 128;
    const int nBase  = blockIdx.y * 16;

    // staging roles
    const int row = tid & 127;           // A: voxel row
    const int chB = (tid >> 7) << 4;     // A: channel half (0/16)
    const int nB  = tid >> 4;            // B: column 0..15
    const int kp  = (tid & 15) << 1;     // B: even k 0..30
    const int vS  = blockM + row;
    const int dS = vS / (Dq * Dq), hS = (vS / Dq) % Dq, wS = vS % Dq;

    const int PAD = (STRIDE == 1) ? (KS >> 1) : 0;
    const int kChunks = (Cin + 31) >> 5;

    v8f acc = {};
    int it = 0;
    for (int td = 0; td < KS; ++td)
    for (int th = 0; th < KS; ++th)
    for (int tw = 0; tw < KS; ++tw) {
        const int id = dS * STRIDE + td - PAD;
        const int ih = hS * STRIDE + th - PAD;
        const int iw = wS * STRIDE + tw - PAD;
        const bool inb = (id >= 0) & (id < Din) & (ih >= 0) & (ih < Din) &
                         (iw >= 0) & (iw < Din);
        const float inbF = inb ? 1.f : 0.f;  // v_cndmask, no branch
        const int idc = clampi(id, 0, Din - 1);
        const int ihc = clampi(ih, 0, Din - 1);
        const int iwc = clampi(iw, 0, Din - 1);
        const float* aPtr = in + ((idc * Din + ihc) * Din + iwc) * Cin;
        const float* wTap = w + (((td * KS) + th) * KS + tw) * Cin * Cout;

        for (int kc = 0; kc < kChunks; ++kc, ++it) {
            const int buf = it & 1;
            const int kb = kc << 5;
            // ---- stage A: 16 channels of one voxel row, packed pairs ----
            {
                const int c0 = kb + chB;
                unsigned t[8];
                if (c0 + 16 <= Cin) {  // full chunk: 4x global_load_b128
                    const float4* p = (const float4*)(aPtr + c0);
#pragma unroll
                    for (int q = 0; q < 4; ++q) {
                        float4 v4 = p[q];
                        t[2 * q]     = pack2bf(v4.x * inbF, v4.y * inbF);
                        t[2 * q + 1] = pack2bf(v4.z * inbF, v4.w * inbF);
                    }
                } else {  // tail: clamped float2 pairs (Cin even >= 2)
#pragma unroll
                    for (int j = 0; j < 8; ++j) {
                        const int k0 = c0 + 2 * j;
                        const int kcl = (k0 + 1 < Cin) ? k0 : (Cin - 2);
                        v2f p2 = *(const v2f*)(aPtr + kcl);
                        const float m0 = (k0     < Cin) ? inbF : 0.f;
                        const float m1 = (k0 + 1 < Cin) ? inbF : 0.f;
                        t[j] = pack2bf(p2[0] * m0, p2[1] * m1);
                    }
                }
                uint4* dstA = (uint4*)(ldsA[buf] + row * 32 + chB);
                dstA[0] = make_uint4(t[0], t[1], t[2], t[3]);
                dstA[1] = make_uint4(t[4], t[5], t[6], t[7]);
            }
            // ---- stage B transposed [n][k]: one packed pair per thread ----
            {
                const int gk0 = kb + kp, gk1 = gk0 + 1, gn = nBase + nB;
                const float bm0 = ((gk0 < Cin) && (gn < Cout)) ? 1.f : 0.f;
                const float bm1 = ((gk1 < Cin) && (gn < Cout)) ? 1.f : 0.f;
                const int gnc = (gn < Cout) ? gn : 0;
                const int i0 = ((gk0 < Cin) ? gk0 : 0) * Cout + gnc;
                const int i1 = ((gk1 < Cin) ? gk1 : 0) * Cout + gnc;
                *(unsigned*)(ldsB[buf] + nB * 32 + kp) =
                    pack2bf(wTap[i0] * bm0, wTap[i1] * bm1);
            }
            __syncthreads();  // single barrier per K-step (double-buffered)
            // ---- fragments (4x ds_load_b128) + WMMA ----
            v16bf afrag = frag_from_lds(ldsA[buf] + (wave * 16 + lr) * 32, hi);
            v16bf bfrag = frag_from_lds(ldsB[buf] + lr * 32, hi);
            acc = __builtin_amdgcn_wmma_f32_16x16x32_bf16(
                false, afrag, false, bfrag, (short)0, acc, false, false);
        }
    }

    // ---- epilogue: bias, mask, relu, residual, channel-slice store ----
    const int n = nBase + lr;
    if (n < Cout) {
        const float bv = bias[n];
        const int mBase = blockM + wave * 16 + hi * 8;
#pragma unroll
        for (int r = 0; r < 8; ++r) {
            const int m = mBase + r;  // C layout: VGPR r -> M=r (+8 for hi)
            float val = (acc[r] + bv) * mask[m];
            if (RELU) val = fmaxf(val, 0.f);
            if (residual) val += residual[m * resCtot + resCoff + n];
            out[m * outCtot + outCoff + n] = val;
        }
    }
}

// ---------------------------------------------------------------------------
// Generative transposed conv k=2,s=2: each fine voxel has exactly one coarse
// source + one of 8 taps -> parity GEMM (grid.z = parity).  Same LDS-staged
// WMMA structure; fused relu+mask; writes low half of the concat buffer.
// ---------------------------------------------------------------------------
__global__ __launch_bounds__(256) void upconv2x_wmma_kernel(
    const float* __restrict__ in, const float* __restrict__ w,
    const float* __restrict__ bias, const float* __restrict__ mask,
    float* __restrict__ out, int Dc, int Cin, int Cout,
    int outCtot, int outCoff)
{
    __shared__ __align__(16) bf16_t ldsA[2][128 * 32];
    __shared__ __align__(16) bf16_t ldsB[2][16 * 32];

    const int tid  = threadIdx.x;
    const int lane = tid & 31;
    const int wave = tid >> 5;
    const int hi   = lane >> 4;
    const int lr   = lane & 15;
    const int blockM = blockIdx.x * 128;
    const int nBase  = blockIdx.y * 16;
    const int tap = blockIdx.z;
    const int pd = (tap >> 2) & 1, ph = (tap >> 1) & 1, pw = tap & 1;
    const float* wTap = w + tap * Cin * Cout;

    const int row = tid & 127;
    const int chB = (tid >> 7) << 4;
    const int nB  = tid >> 4;
    const int kp  = (tid & 15) << 1;
    const float* aPtr = in + (blockM + row) * Cin;

    const int kChunks = (Cin + 31) >> 5;
    v8f acc = {};
    for (int kc = 0; kc < kChunks; ++kc) {
        const int buf = kc & 1;
        const int kb = kc << 5;
        {
            const int c0 = kb + chB;
            unsigned t[8];
            if (c0 + 16 <= Cin) {
                const float4* p = (const float4*)(aPtr + c0);
#pragma unroll
                for (int q = 0; q < 4; ++q) {
                    float4 v4 = p[q];
                    t[2 * q]     = pack2bf(v4.x, v4.y);
                    t[2 * q + 1] = pack2bf(v4.z, v4.w);
                }
            } else {  // tail: clamped float2 pairs (Cin even >= 2)
#pragma unroll
                for (int j = 0; j < 8; ++j) {
                    const int k0 = c0 + 2 * j;
                    const int kcl = (k0 + 1 < Cin) ? k0 : (Cin - 2);
                    v2f p2 = *(const v2f*)(aPtr + kcl);
                    const float m0 = (k0     < Cin) ? 1.f : 0.f;
                    const float m1 = (k0 + 1 < Cin) ? 1.f : 0.f;
                    t[j] = pack2bf(p2[0] * m0, p2[1] * m1);
                }
            }
            uint4* dstA = (uint4*)(ldsA[buf] + row * 32 + chB);
            dstA[0] = make_uint4(t[0], t[1], t[2], t[3]);
            dstA[1] = make_uint4(t[4], t[5], t[6], t[7]);
        }
        {
            const int gk0 = kb + kp, gk1 = gk0 + 1, gn = nBase + nB;
            const float bm0 = ((gk0 < Cin) && (gn < Cout)) ? 1.f : 0.f;
            const float bm1 = ((gk1 < Cin) && (gn < Cout)) ? 1.f : 0.f;
            const int gnc = (gn < Cout) ? gn : 0;
            const int i0 = ((gk0 < Cin) ? gk0 : 0) * Cout + gnc;
            const int i1 = ((gk1 < Cin) ? gk1 : 0) * Cout + gnc;
            *(unsigned*)(ldsB[buf] + nB * 32 + kp) =
                pack2bf(wTap[i0] * bm0, wTap[i1] * bm1);
        }
        __syncthreads();
        v16bf afrag = frag_from_lds(ldsA[buf] + (wave * 16 + lr) * 32, hi);
        v16bf bfrag = frag_from_lds(ldsB[buf] + lr * 32, hi);
        acc = __builtin_amdgcn_wmma_f32_16x16x32_bf16(
            false, afrag, false, bfrag, (short)0, acc, false, false);
    }

    const int n = nBase + lr;
    if (n < Cout) {
        const float bv = bias[n];
        const int Df = 2 * Dc;
        const int mBase = blockM + wave * 16 + hi * 8;
#pragma unroll
        for (int r = 0; r < 8; ++r) {
            const int m = mBase + r;
            const int d = m / (Dc * Dc), h = (m / Dc) % Dc, ww = m % Dc;
            const int fi = ((2 * d + pd) * Df + (2 * h + ph)) * Df + (2 * ww + pw);
            float val = fmaxf((acc[r] + bv) * mask[fi], 0.f);
            out[fi * outCtot + outCoff + n] = val;
        }
    }
}

// --------------------------- elementwise helpers ---------------------------
__global__ void mask_from_x_kernel(const float* __restrict__ x,
                                   float* __restrict__ m, int n) {
    int i = blockIdx.x * blockDim.x + threadIdx.x;
    if (i < n) m[i] = (x[i] != 0.f) ? 1.f : 0.f;
}

__global__ void pool_mask_kernel(const float* __restrict__ mf,
                                 float* __restrict__ mc, int Dc) {
    int v = blockIdx.x * blockDim.x + threadIdx.x;
    int nv = Dc * Dc * Dc;
    if (v >= nv) return;
    int Df = 2 * Dc;
    int d = v / (Dc * Dc), h = (v / Dc) % Dc, w = v % Dc;
    float m = 0.f;
    for (int a = 0; a < 2; ++a)
        for (int b = 0; b < 2; ++b)
            for (int c = 0; c < 2; ++c)
                m = fmaxf(m, mf[((2 * d + a) * Df + 2 * h + b) * Df + 2 * w + c]);
    mc[v] = m;
}

__global__ void copy_channels_kernel(const float* __restrict__ in,
                                     float* __restrict__ out, int numVox,
                                     int C, int outCtot, int outCoff) {
    int i = blockIdx.x * blockDim.x + threadIdx.x;
    if (i >= numVox * C) return;
    int v = i / C, c = i % C;
    out[v * outCtot + outCoff + c] = in[i];
}

// zero-pad 1 -> 2 channels (activations)
__global__ void pad_x2_kernel(const float* __restrict__ x,
                              float* __restrict__ xp, int n) {
    int i = blockIdx.x * blockDim.x + threadIdx.x;
    if (i < n) {
        xp[2 * i]     = x[i];
        xp[2 * i + 1] = 0.f;
    }
}

// zero-pad conv0 weights [taps][1][Cout] -> [taps][2][Cout]
__global__ void pad_w2_kernel(const float* __restrict__ w,
                              float* __restrict__ wp, int taps, int Cout) {
    int i = blockIdx.x * blockDim.x + threadIdx.x;
    int total = taps * 2 * Cout;
    if (i >= total) return;
    int n = i % Cout;
    int k = (i / Cout) & 1;
    int t = i / (2 * Cout);
    wp[i] = (k == 0) ? w[t * Cout + n] : 0.f;
}

// ------------------------------ host side ----------------------------------
namespace {

struct WBp { const float* w; const float* b; };

void conv_go(hipStream_t s, const float* in, const float* w, const float* b,
             const float* mask, const float* res, float* out,
             int Din, int Dq, int Cin, int Cout, int KS, int STRIDE, bool relu,
             int outCtot, int outCoff, int resCtot, int resCoff) {
    int numVox = Dq * Dq * Dq;           // always a multiple of 128 here
    dim3 grid(numVox / 128, (Cout + 15) / 16, 1);
    dim3 block(256, 1, 1);
    if (KS == 3) {
        if (relu) conv3d_wmma_kernel<3, 1, true><<<grid, block, 0, s>>>(
            in, w, b, mask, res, out, Din, Cin, Dq, Cout, outCtot, outCoff, resCtot, resCoff);
        else conv3d_wmma_kernel<3, 1, false><<<grid, block, 0, s>>>(
            in, w, b, mask, res, out, Din, Cin, Dq, Cout, outCtot, outCoff, resCtot, resCoff);
    } else if (KS == 2) {
        conv3d_wmma_kernel<2, 2, true><<<grid, block, 0, s>>>(
            in, w, b, mask, res, out, Din, Cin, Dq, Cout, outCtot, outCoff, resCtot, resCoff);
    } else {
        if (relu) conv3d_wmma_kernel<1, 1, true><<<grid, block, 0, s>>>(
            in, w, b, mask, res, out, Din, Cin, Dq, Cout, outCtot, outCoff, resCtot, resCoff);
        else conv3d_wmma_kernel<1, 1, false><<<grid, block, 0, s>>>(
            in, w, b, mask, res, out, Din, Cin, Dq, Cout, outCtot, outCoff, resCtot, resCoff);
    }
}

void upconv_go(hipStream_t s, const float* in, const float* w, const float* b,
               const float* mask, float* out, int Dc, int Cin, int Cout,
               int outCtot, int outCoff) {
    int numVox = Dc * Dc * Dc;
    dim3 grid(numVox / 128, (Cout + 15) / 16, 8);
    dim3 block(256, 1, 1);
    upconv2x_wmma_kernel<<<grid, block, 0, s>>>(in, w, b, mask, out, Dc, Cin,
                                                Cout, outCtot, outCoff);
}

// inverted residual block: y = concat(o0,o1) + x  (fused via slice+residual)
void irb_go(hipStream_t s, const float* x, const WBp* p, const float* mask,
            float* y, float* t0, float* t1, int D, int C) {
    int C4 = C / 4, C2 = C / 2;
    conv_go(s, x,  p[0].w, p[0].b, mask, nullptr, t0, D, D, C,  C4, 3, 1, true,  C4, 0,  0, 0);
    conv_go(s, t0, p[1].w, p[1].b, mask, x,       y,  D, D, C4, C2, 3, 1, false, C,  0,  C, 0);
    conv_go(s, x,  p[2].w, p[2].b, mask, nullptr, t0, D, D, C,  C4, 1, 1, true,  C4, 0,  0, 0);
    conv_go(s, t0, p[3].w, p[3].b, mask, nullptr, t1, D, D, C4, C4, 3, 1, true,  C4, 0,  0, 0);
    conv_go(s, t1, p[4].w, p[4].b, mask, nullptr, y,  D, D, C4, C2, 1, 1, false, C, C2,  C, C2);
}

void blocks_go(hipStream_t s, int& wi, void* const* d_in, const float* in,
               float* pA, float* pB, float* dst, const float* mask, float* t0,
               float* t1, int D, int C) {
    WBp p[5];
    auto rd = [&]() {
        for (int i = 0; i < 5; ++i) {
            p[i].w = (const float*)d_in[wi++];
            p[i].b = (const float*)d_in[wi++];
        }
    };
    rd(); irb_go(s, in, p, mask, pA, t0, t1, D, C);
    rd(); irb_go(s, pA, p, mask, pB, t0, t1, D, C);
    rd(); irb_go(s, pB, p, mask, dst, t0, t1, D, C);
}

}  // namespace

extern "C" void kernel_launch(void* const* d_in, const int* in_sizes, int n_in,
                              void* d_out, int out_size, void* d_ws,
                              size_t ws_size, hipStream_t stream) {
    (void)in_sizes; (void)n_in; (void)out_size; (void)ws_size;
    const float* x = (const float*)d_in[0];
    float* dout = (float*)d_out;

    const int D0 = 64, D1 = 32, D2 = 16, D3 = 8;
    const int R0 = D0 * D0 * D0, R1 = D1 * D1 * D1, R2 = D2 * D2 * D2,
              R3 = D3 * D3 * D3;

    // bump allocator over workspace (floats)
    size_t off = 0;
    auto alloc = [&](size_t n) { float* p = (float*)d_ws + off; off += n; return p; };
    float* M0 = alloc(R0); float* M1 = alloc(R1);
    float* M2 = alloc(R2); float* M3 = alloc(R3);
    float* XPAD = alloc((size_t)R0 * 2);                  // conv0 input, 2ch
    float* WPAD = alloc(27 * 2 * 16);                     // conv0 weights, 2ch
    float* OUTX = alloc((size_t)R0 * 16);                 // encoder skip @R0
    float* X0a = alloc((size_t)R0 * 16); float* X0b = alloc((size_t)R0 * 16);
    float* CAT0 = alloc((size_t)R0 * 32);
    float* T0a = alloc((size_t)R0 * 4);  float* T0b = alloc((size_t)R0 * 4);
    float* OUT0 = alloc((size_t)R1 * 32);                 // encoder skip @R1
    float* X1a = alloc((size_t)R1 * 32); float* X1b = alloc((size_t)R1 * 32);
    float* CAT1 = alloc((size_t)R1 * 64);
    float* T1a = alloc((size_t)R1 * 8);  float* T1b = alloc((size_t)R1 * 8);
    float* OUT1 = alloc((size_t)R2 * 64);                 // encoder skip @R2
    float* X2a = alloc((size_t)R2 * 64); float* X2b = alloc((size_t)R2 * 64);
    float* CAT2 = alloc((size_t)R2 * 128);
    float* T2a = alloc((size_t)R2 * 16); float* T2b = alloc((size_t)R2 * 16);
    float* X3a = alloc((size_t)R3 * 32); float* X3b = alloc((size_t)R3 * 32);
    float* T3a = alloc((size_t)R3 * 8);  float* T3b = alloc((size_t)R3 * 8);
    float* OUT2 = alloc((size_t)R3 * 8);

    // masks: m0 = (x != 0) (x = normal * mask -> a.s. exact); pool 3x
    mask_from_x_kernel<<<(R0 + 255) / 256, 256, 0, stream>>>(x, M0, R0);
    pool_mask_kernel<<<(R1 + 255) / 256, 256, 0, stream>>>(M0, M1, D1);
    pool_mask_kernel<<<(R2 + 255) / 256, 256, 0, stream>>>(M1, M2, D2);
    pool_mask_kernel<<<(R3 + 255) / 256, 256, 0, stream>>>(M2, M3, D3);

    // weight iterator in dict-insertion order
    int wi = 2;
    auto nw = [&](WBp& o) {
        o.w = (const float*)d_in[wi++];
        o.b = (const float*)d_in[wi++];
    };
    WBp wb;

    // ---- encoder ----
    nw(wb);  // conv0: pad 1->2 channels, then conv with Cin=2
    pad_x2_kernel<<<(R0 + 255) / 256, 256, 0, stream>>>(x, XPAD, R0);
    pad_w2_kernel<<<(27 * 2 * 16 + 255) / 256, 256, 0, stream>>>(wb.w, WPAD, 27, 16);
    conv_go(stream, XPAD, WPAD, wb.b, M0, nullptr, OUTX, D0, D0, 2, 16, 3, 1, true, 16, 0, 0, 0);
    nw(wb);  // down0
    conv_go(stream, OUTX, wb.w, wb.b, M1, nullptr, X1a, D0, D1, 16, 32, 2, 2, true, 32, 0, 0, 0);
    blocks_go(stream, wi, d_in, X1a, X1b, X1a, OUT0, M1, T1a, T1b, D1, 32);  // block0

    nw(wb);  // conv1
    conv_go(stream, OUT0, wb.w, wb.b, M1, nullptr, X1a, D1, D1, 32, 32, 3, 1, true, 32, 0, 0, 0);
    nw(wb);  // down1
    conv_go(stream, X1a, wb.w, wb.b, M2, nullptr, X2a, D1, D2, 32, 64, 2, 2, true, 64, 0, 0, 0);
    blocks_go(stream, wi, d_in, X2a, X2b, X2a, OUT1, M2, T2a, T2b, D2, 64);  // block1

    nw(wb);  // conv2
    conv_go(stream, OUT1, wb.w, wb.b, M2, nullptr, X2a, D2, D2, 64, 64, 3, 1, true, 64, 0, 0, 0);
    nw(wb);  // down2
    conv_go(stream, X2a, wb.w, wb.b, M3, nullptr, X3a, D2, D3, 64, 32, 2, 2, true, 32, 0, 0, 0);
    blocks_go(stream, wi, d_in, X3a, X3b, X3a, X3b, M3, T3a, T3b, D3, 32);   // block2

    nw(wb);  // conv3 (no relu)
    conv_go(stream, X3b, wb.w, wb.b, M3, nullptr, OUT2, D3, D3, 32, 8, 3, 1, false, 8, 0, 0, 0);

    // ---- decoder ----
    nw(wb);  // up0
    upconv_go(stream, OUT2, wb.w, wb.b, M2, CAT2, D3, 8, 64, 128, 0);
    copy_channels_kernel<<<((size_t)R2 * 64 + 255) / 256, 256, 0, stream>>>(
        OUT1, CAT2, R2, 64, 128, 64);
    nw(wb);  // conv_0
    conv_go(stream, CAT2, wb.w, wb.b, M2, nullptr, X2a, D2, D2, 128, 64, 3, 1, true, 64, 0, 0, 0);
    blocks_go(stream, wi, d_in, X2a, X2b, X2a, X2b, M2, T2a, T2b, D2, 64);   // block_0

    nw(wb);  // up1
    upconv_go(stream, X2b, wb.w, wb.b, M1, CAT1, D2, 64, 32, 64, 0);
    copy_channels_kernel<<<((size_t)R1 * 32 + 255) / 256, 256, 0, stream>>>(
        OUT0, CAT1, R1, 32, 64, 32);
    nw(wb);  // conv_1
    conv_go(stream, CAT1, wb.w, wb.b, M1, nullptr, X1a, D1, D1, 64, 32, 3, 1, true, 32, 0, 0, 0);
    blocks_go(stream, wi, d_in, X1a, X1b, X1a, X1b, M1, T1a, T1b, D1, 32);   // block_1

    nw(wb);  // up2
    upconv_go(stream, X1b, wb.w, wb.b, M0, CAT0, D1, 32, 16, 32, 0);
    copy_channels_kernel<<<((size_t)R0 * 16 + 255) / 256, 256, 0, stream>>>(
        OUTX, CAT0, R0, 16, 32, 16);
    nw(wb);  // conv_2
    conv_go(stream, CAT0, wb.w, wb.b, M0, nullptr, X0a, D0, D0, 32, 16, 3, 1, true, 16, 0, 0, 0);
    blocks_go(stream, wi, d_in, X0a, X0b, X0a, X0b, M0, T0a, T0b, D0, 16);   // block_2

    nw(wb);  // conv_3 (no relu) -> final output, 64^3 x 32
    conv_go(stream, X0b, wb.w, wb.b, M0, nullptr, dout, D0, D0, 16, 32, 3, 1, false, 32, 0, 0, 0);
}
